// GAT_70231305224956
// MI455X (gfx1250) — compile-verified, matrix operand order checked by reading, not run on hardware
//
#include <hip/hip_runtime.h>
#include <hip/hip_bf16.h>

// ---------------- types ----------------
typedef __attribute__((ext_vector_type(16))) __bf16 bfvec16;
typedef __attribute__((ext_vector_type(8)))  __bf16 bfvec8;
typedef __attribute__((ext_vector_type(8)))  float  fvec8;

#define N_NODES 50000
#define N_EDGES 800000
#define KDIM    256
#define KT_CT   (KDIM / 32)          // 8 k-tiles
#define HEADS   4

__device__ __forceinline__ unsigned short f2bf(float f) {
    unsigned int u = __float_as_uint(f);
    unsigned int r = u + 0x7FFFu + ((u >> 16) & 1u);   // round-to-nearest-even
    return (unsigned short)(r >> 16);
}
__device__ __forceinline__ float bf2f(unsigned short u) {
    return __uint_as_float(((unsigned int)u) << 16);
}
__device__ __forceinline__ void atomicMaxF(float* addr, float val) {
    unsigned int* ua = (unsigned int*)addr;
    unsigned int old = *ua;
    while (true) {
        float cur = __uint_as_float(old);
        if (cur >= val) break;
        unsigned int assumed = old;
        old = atomicCAS(ua, assumed, __float_as_uint(val));
        if (old == assumed) break;
    }
}

// ---------------- fill ----------------
__global__ void fill_f32(float* __restrict__ p, float v, long n) {
    long i = (long)blockIdx.x * blockDim.x + threadIdx.x;
    if (i < n) p[i] = v;
}

// ---------------- fp32 -> bf16 ----------------
__global__ void f32_to_bf16(const float* __restrict__ src, unsigned short* __restrict__ dst, long n) {
    long i = (long)blockIdx.x * blockDim.x + threadIdx.x;
    if (i < n) dst[i] = f2bf(src[i]);
}

// ---- pack W [K x Ncols] row-major fp32 -> per-lane WMMA B fragments (bf16) ----
// Fragment for (nt, kt): lane l holds 16 contiguous bf16; element e maps to
// k = kt*32 + ((e<8 ? e : e+8) + (l>>4)*8), n = nt*16 + (l&15).
__global__ void pack_w_bf16(const float* __restrict__ W, unsigned short* __restrict__ Bp,
                            int Ncols, int Np) {
    int idx = blockIdx.x * blockDim.x + threadIdx.x;
    int total = KT_CT * (Np >> 4) * 512;
    if (idx >= total) return;
    int e    = idx & 15;
    int lane = (idx >> 4) & 31;
    int tile = idx >> 9;
    int kt = tile % KT_CT;
    int nt = tile / KT_CT;
    int hi8 = (lane >> 4) * 8;
    int n  = nt * 16 + (lane & 15);
    int kk = (e < 8 ? e : e + 8) + hi8;
    int k  = kt * 32 + kk;
    float v = (n < Ncols) ? W[(size_t)k * Ncols + n] : 0.0f;
    Bp[idx] = f2bf(v);
}

// ---------------- WMMA GEMM: C[M x Np] = A[M x 256](bf16) * Bpacked + (acc?C:0) ----
// 256 threads = 8 waves; block tile 64(M) x 64(N); each wave owns 16x32.
// B slab for the block (64 cols x 256 K = 32KB) is staged into LDS with the
// CDNA5 async global->LDS path, then WMMA reads B via ds loads. A fragments
// are software-pipelined (kt+1 loads issued before kt's WMMAs).
__global__ __launch_bounds__(256) void gemm_bf16_wmma(
    const unsigned short* __restrict__ A, const unsigned short* __restrict__ Bp,
    float* __restrict__ C, int M, int Np, int acc) {
    __shared__ unsigned short ldsB[4 * KT_CT * 512];   // 32 KB
    const int tid   = threadIdx.x;
    const int lane  = tid & 31;
    const int wave  = tid >> 5;
    const int waveM = wave & 3;                 // 4 waves along M
    const int waveN = wave >> 2;                // 2 waves along N (32 cols each)
    const int rowBase = blockIdx.x * 64 + waveM * 16;
    const int colBase = blockIdx.y * 64 + waveN * 32;

    // ---- async stage B slab (contiguous in packed layout) ----
    {
        const size_t slab = (size_t)(blockIdx.y * 4) * KT_CT * 512;   // elements
        unsigned long long g = (unsigned long long)(const void*)(Bp + slab) + (size_t)tid * 16;
        unsigned int l = ((unsigned int)(size_t)(&ldsB[0])) + (unsigned int)tid * 16;
#pragma unroll
        for (int p = 0; p < (4 * KT_CT * 512 * 2) / (256 * 16); ++p) {  // 8 passes
            asm volatile("global_load_async_to_lds_b128 %0, %1, off"
                         :: "v"(l), "v"(g) : "memory");
            g += 256 * 16;
            l += 256 * 16;
        }
        asm volatile("s_wait_asynccnt 0x0" ::: "memory");
    }
    __syncthreads();

    const int hi8 = (lane >> 4) * 8;
    int mrow = rowBase + (lane & 15);
    if (mrow >= M) mrow = M - 1;                // clamp (stores are guarded)
    const unsigned short* arow = A + (size_t)mrow * KDIM + hi8;
    const unsigned short* lb0 = &ldsB[(size_t)(waveN * 2 + 0) * KT_CT * 512 + lane * 16];
    const unsigned short* lb1 = &ldsB[(size_t)(waveN * 2 + 1) * KT_CT * 512 + lane * 16];

    fvec8 c0 = {}, c1 = {};
    if (acc) {
#pragma unroll
        for (int r = 0; r < 8; ++r) {
            int m = rowBase + r + (lane >> 4) * 8;
            if (m < M) {
                c0[r] = C[(size_t)m * Np + colBase + (lane & 15)];
                c1[r] = C[(size_t)m * Np + colBase + 16 + (lane & 15)];
            }
        }
    }

    bfvec8 alo = *(const bfvec8*)(arow);
    bfvec8 ahi = *(const bfvec8*)(arow + 16);
#pragma unroll
    for (int kt = 0; kt < KT_CT; ++kt) {
        bfvec8 nlo = {}, nhi = {};
        if (kt + 1 < KT_CT) {                       // issue next A tile early
            const unsigned short* ap = arow + (kt + 1) * 32;
            nlo = *(const bfvec8*)(ap);
            nhi = *(const bfvec8*)(ap + 16);
        }
        bfvec16 af;
#pragma unroll
        for (int e = 0; e < 8; ++e) { af[e] = alo[e]; af[e + 8] = ahi[e]; }
        bfvec16 bf0 = *(const bfvec16*)(lb0 + kt * 512);
        bfvec16 bf1 = *(const bfvec16*)(lb1 + kt * 512);
        c0 = __builtin_amdgcn_wmma_f32_16x16x32_bf16(false, af, false, bf0, (short)0, c0, false, false);
        c1 = __builtin_amdgcn_wmma_f32_16x16x32_bf16(false, af, false, bf1, (short)0, c1, false, false);
        alo = nlo; ahi = nhi;
    }

#pragma unroll
    for (int r = 0; r < 8; ++r) {
        int m = rowBase + r + (lane >> 4) * 8;
        if (m < M) {
            C[(size_t)m * Np + colBase + (lane & 15)]      = c0[r];
            C[(size_t)m * Np + colBase + 16 + (lane & 15)] = c1[r];
        }
    }
}

// ---------------- attention coefficients el/er ----------------
__global__ void attn_lr(const float* __restrict__ feat, const float* __restrict__ al,
                        const float* __restrict__ ar, float* __restrict__ el,
                        float* __restrict__ er, int N, int D, int ld) {
    int t = blockIdx.x * blockDim.x + threadIdx.x;
    if (t >= N * HEADS) return;
    int n = t >> 2, h = t & 3;
    const float* f  = feat + (size_t)n * ld + h * D;
    const float* a1 = al + h * D;
    const float* a2 = ar + h * D;
    float s1 = 0.f, s2 = 0.f;
    for (int d = 0; d < D; ++d) { float v = f[d]; s1 += v * a1[d]; s2 += v * a2[d]; }
    el[t] = s1; er[t] = s2;
}

// ---------------- edge pass A: segment max ----------------
__global__ void edge_max(const int* __restrict__ src, const int* __restrict__ dst,
                         const float* __restrict__ el, const float* __restrict__ er,
                         float* __restrict__ mx, long EH) {
    long t = (long)blockIdx.x * blockDim.x + threadIdx.x;
    if (t >= EH) return;
    int e = (int)(t >> 2), h = (int)(t & 3);
    int s = src[e], d = dst[e];
    float v = el[s * HEADS + h] + er[d * HEADS + h];
    v = v > 0.f ? v : 0.2f * v;
    atomicMaxF(&mx[d * HEADS + h], v);
}

// ---------------- edge pass B: segment sum of exp ----------------
__global__ void edge_sum(const int* __restrict__ src, const int* __restrict__ dst,
                         const float* __restrict__ el, const float* __restrict__ er,
                         const float* __restrict__ mx, float* __restrict__ sm, long EH) {
    long t = (long)blockIdx.x * blockDim.x + threadIdx.x;
    if (t >= EH) return;
    int e = (int)(t >> 2), h = (int)(t & 3);
    int s = src[e], d = dst[e];
    float v = el[s * HEADS + h] + er[d * HEADS + h];
    v = v > 0.f ? v : 0.2f * v;
    atomicAdd(&sm[d * HEADS + h], __expf(v - mx[d * HEADS + h]));
}

// ---------------- edge pass C: alpha-weighted scatter (64 threads / edge) ----------------
__global__ void edge_scatter(const int* __restrict__ src, const int* __restrict__ dst,
                             const float* __restrict__ el, const float* __restrict__ er,
                             const float* __restrict__ mx, const float* __restrict__ sm,
                             const float* __restrict__ feat, float* __restrict__ rst,
                             int D, int ld) {
    long t = (long)blockIdx.x * blockDim.x + threadIdx.x;
    int e = (int)(t >> 6);
    if (e >= N_EDGES) return;
    int r = (int)(t & 63);
    int h = r >> 4, i = r & 15;
    int sN = src[e], dN = dst[e];
    float v = el[sN * HEADS + h] + er[dN * HEADS + h];
    v = v > 0.f ? v : 0.2f * v;
    float alpha = __expf(v - mx[dN * HEADS + h]) / sm[dN * HEADS + h];
    const float* fs = feat + (size_t)sN * ld + h * D;
    float*       rd = rst  + (size_t)dN * ld + h * D;
    for (int d = i; d < D; d += 16) atomicAdd(&rd[d], alpha * fs[d]);
}

// ---------------- finalize (layers 0/1): bias + residual + ELU -> bf16 in place ----------------
__global__ void finalize_act(const float* __restrict__ rst, const float* __restrict__ b,
                             unsigned short* __restrict__ abf, long n, int F, int useRes) {
    long i = (long)blockIdx.x * blockDim.x + threadIdx.x;
    if (i >= n) return;
    float v = rst[i] + b[(int)(i % F)];
    if (useRes) v += bf2f(abf[i]);
    v = v > 0.f ? v : (__expf(v) - 1.f);
    abf[i] = f2bf(v);
}

// ---------------- final: per-head bias + mean over heads ----------------
__global__ void final_mean(const float* __restrict__ rst, const float* __restrict__ b2,
                           float* __restrict__ out, int N) {
    int t = blockIdx.x * blockDim.x + threadIdx.x;
    if (t >= N * 47) return;
    int n = t / 47, c = t % 47;
    float acc = 0.f;
#pragma unroll
    for (int h = 0; h < HEADS; ++h) acc += rst[(size_t)n * 192 + h * 47 + c] + b2[h * 47 + c];
    out[t] = acc * 0.25f;
}

// ---------------- host launch ----------------
extern "C" void kernel_launch(void* const* d_in, const int* in_sizes, int n_in,
                              void* d_out, int out_size, void* d_ws, size_t ws_size,
                              hipStream_t stream) {
    (void)in_sizes; (void)n_in; (void)out_size; (void)ws_size;
    const float* x   = (const float*)d_in[0];
    const int*   src = (const int*)d_in[1];
    const int*   dst = (const int*)d_in[2];
    const float* W0  = (const float*)d_in[3];
    const float* al0 = (const float*)d_in[4];
    const float* ar0 = (const float*)d_in[5];
    const float* b0  = (const float*)d_in[6];
    const float* W1  = (const float*)d_in[7];
    const float* al1 = (const float*)d_in[8];
    const float* ar1 = (const float*)d_in[9];
    const float* b1  = (const float*)d_in[10];
    const float* W2  = (const float*)d_in[11];
    const float* al2 = (const float*)d_in[12];
    const float* ar2 = (const float*)d_in[13];
    const float* rW2 = (const float*)d_in[14];
    const float* b2  = (const float*)d_in[15];
    float* out = (float*)d_out;

    // workspace carve-up (256-byte aligned)
    char* ws = (char*)d_ws;
    size_t off = 0;
    auto carve = [&](size_t bytes) -> void* {
        void* p = ws + off;
        off = (off + bytes + 255) & ~(size_t)255;
        return p;
    };
    unsigned short* abf = (unsigned short*)carve((size_t)N_NODES * KDIM * 2);  // bf16 activations
    unsigned short* wpA = (unsigned short*)carve((size_t)KDIM * 256 * 2);      // packed weights A
    unsigned short* wpB = (unsigned short*)carve((size_t)KDIM * 256 * 2);      // packed weights B
    float* feat = (float*)carve((size_t)N_NODES * KDIM * 4);
    float* rst  = (float*)carve((size_t)N_NODES * KDIM * 4);
    float* el   = (float*)carve((size_t)N_NODES * HEADS * 4);
    float* er   = (float*)carve((size_t)N_NODES * HEADS * 4);
    float* mx   = (float*)carve((size_t)N_NODES * HEADS * 4);
    float* sm   = (float*)carve((size_t)N_NODES * HEADS * 4);

    const long NH  = (long)N_NODES * HEADS;
    const long EH  = (long)N_EDGES * HEADS;
    const long E64 = (long)N_EDGES * 64;

    auto blocks = [](long n, int t) { return (unsigned)((n + t - 1) / t); };

    // --- input to bf16 ---
    f32_to_bf16<<<blocks((long)N_NODES * KDIM, 256), 256, 0, stream>>>(x, abf, (long)N_NODES * KDIM);

    // === layers 0 and 1 (Np = 256, D = 64) ===
    const float* Ws[2]  = {W0, W1};
    const float* als[2] = {al0, al1};
    const float* ars[2] = {ar0, ar1};
    const float* bs[2]  = {b0, b1};
    for (int L = 0; L < 2; ++L) {
        pack_w_bf16<<<blocks((long)KDIM * 256, 256), 256, 0, stream>>>(Ws[L], wpA, 256, 256);
        dim3 gg((N_NODES + 63) / 64, 256 / 64);
        gemm_bf16_wmma<<<gg, 256, 0, stream>>>(abf, wpA, feat, N_NODES, 256, 0);
        attn_lr<<<blocks(NH, 256), 256, 0, stream>>>(feat, als[L], ars[L], el, er, N_NODES, 64, 256);
        fill_f32<<<blocks(NH, 256), 256, 0, stream>>>(mx, -1e30f, NH);
        fill_f32<<<blocks(NH, 256), 256, 0, stream>>>(sm, 0.f, NH);
        fill_f32<<<blocks((long)N_NODES * 256, 256), 256, 0, stream>>>(rst, 0.f, (long)N_NODES * 256);
        edge_max<<<blocks(EH, 256), 256, 0, stream>>>(src, dst, el, er, mx, EH);
        edge_sum<<<blocks(EH, 256), 256, 0, stream>>>(src, dst, el, er, mx, sm, EH);
        edge_scatter<<<blocks(E64, 256), 256, 0, stream>>>(src, dst, el, er, mx, sm, feat, rst, 64, 256);
        finalize_act<<<blocks((long)N_NODES * 256, 256), 256, 0, stream>>>(
            rst, bs[L], abf, (long)N_NODES * 256, 256, L == 1 ? 1 : 0);
    }

    // === layer 2 (Ncols = 188 -> Np = 192, D = 47, residual = h @ resW2, no act) ===
    pack_w_bf16<<<blocks((long)KDIM * 192, 256), 256, 0, stream>>>(W2, wpA, 188, 192);
    pack_w_bf16<<<blocks((long)KDIM * 192, 256), 256, 0, stream>>>(rW2, wpB, 188, 192);
    {
        dim3 gg((N_NODES + 63) / 64, 192 / 64);
        gemm_bf16_wmma<<<gg, 256, 0, stream>>>(abf, wpA, feat, N_NODES, 192, 0);
        attn_lr<<<blocks(NH, 256), 256, 0, stream>>>(feat, al2, ar2, el, er, N_NODES, 47, 192);
        fill_f32<<<blocks(NH, 256), 256, 0, stream>>>(mx, -1e30f, NH);
        fill_f32<<<blocks(NH, 256), 256, 0, stream>>>(sm, 0.f, NH);
        fill_f32<<<blocks((long)N_NODES * 192, 256), 256, 0, stream>>>(rst, 0.f, (long)N_NODES * 192);
        edge_max<<<blocks(EH, 256), 256, 0, stream>>>(src, dst, el, er, mx, EH);
        edge_sum<<<blocks(EH, 256), 256, 0, stream>>>(src, dst, el, er, mx, sm, EH);
        edge_scatter<<<blocks(E64, 256), 256, 0, stream>>>(src, dst, el, er, mx, sm, feat, rst, 47, 192);
        // residual projection accumulated straight into the aggregation buffer
        gemm_bf16_wmma<<<gg, 256, 0, stream>>>(abf, wpB, rst, N_NODES, 192, 1);
    }
    final_mean<<<blocks((long)N_NODES * 47, 256), 256, 0, stream>>>(rst, b2, out, N_NODES);
}
